// PointNetSetAbstraction_59717225284091
// MI455X (gfx1250) — matrix operand order repaired and checked by reading, not compiled, not run on hardware
//
#include <hip/hip_runtime.h>

// ---------------------------------------------------------------------------
// PointNet++ Set Abstraction for MI455X (gfx1250, wave32, WMMA)
//   B=16, N=4096, C_in=64, npoint M=1024, k=32, mlp = [64, 64, 128]
//   d_out = [ new_xyz (B,M,3) | new_points (B,M,128) ]  (float32)
// ---------------------------------------------------------------------------

typedef __attribute__((ext_vector_type(16))) _Float16 v16h;
typedef __attribute__((ext_vector_type(8)))  _Float16 v8h;
typedef __attribute__((ext_vector_type(8)))  float    v8f;
typedef int v4i_ __attribute__((vector_size(16)));   // matches builtin param

#define B_    16
#define N_    4096
#define M_    1024
#define KNN_  32
#define CIN_  64
#define ROWS_ (B_ * M_ * KNN_)      // 524288 rows through the MLP

// ---- CDNA5 async global->LDS copy (probe via __has_builtin) ----------------
#if defined(__has_builtin)
#if __has_builtin(__builtin_amdgcn_global_load_async_to_lds_b128)
#define HAVE_ASYNC_LDS 1
#endif
#endif

#ifdef HAVE_ASYNC_LDS
__device__ __forceinline__ void async_cp_b128(const _Float16* gsrc, _Float16* ldst) {
  __builtin_amdgcn_global_load_async_to_lds_b128(
      (__attribute__((address_space(1))) v4i_*)gsrc,
      (__attribute__((address_space(3))) v4i_*)ldst, 0, 0);
}
__device__ __forceinline__ void async_wait0() {
#if __has_builtin(__builtin_amdgcn_s_wait_asynccnt)
  __builtin_amdgcn_s_wait_asynccnt(0);
#else
  asm volatile("s_wait_asynccnt 0" ::: "memory");
#endif
}
#endif

// ------------------------------- FPS ---------------------------------------
__global__ __launch_bounds__(1024) void fps_kernel(const float* __restrict__ xyz,
                                                   float* __restrict__ new_xyz) {
  __shared__ float dist[N_];
  __shared__ float rv[32];
  __shared__ int   ri[32];
  __shared__ int   s_far;
  const int b = blockIdx.x;
  const float* X = xyz + (size_t)b * N_ * 3;
  const int tid = threadIdx.x, lane = tid & 31, wv = tid >> 5;
  for (int i = tid; i < N_; i += 1024) dist[i] = 1e10f;
  if (tid == 0) s_far = 0;
  __syncthreads();
  for (int m = 0; m < M_; ++m) {
    int far = s_far;
    if (tid == 0) {
      float* o = new_xyz + ((size_t)b * M_ + m) * 3;
      o[0] = X[far * 3 + 0]; o[1] = X[far * 3 + 1]; o[2] = X[far * 3 + 2];
    }
    float cx = X[far * 3 + 0], cy = X[far * 3 + 1], cz = X[far * 3 + 2];
    float bv = -1.f; int bi = 0;
    for (int i = tid; i < N_; i += 1024) {
      float dx = X[i * 3 + 0] - cx, dy = X[i * 3 + 1] - cy, dz = X[i * 3 + 2] - cz;
      float d = dx * dx + dy * dy + dz * dz;
      float nd = fminf(dist[i], d);
      dist[i] = nd;
      if (nd > bv) { bv = nd; bi = i; }
    }
#pragma unroll
    for (int off = 16; off > 0; off >>= 1) {
      float ov = __shfl_xor(bv, off, 32);
      int   oi = __shfl_xor(bi, off, 32);
      if (ov > bv || (ov == bv && oi < bi)) { bv = ov; bi = oi; }
    }
    if (lane == 0) { rv[wv] = bv; ri[wv] = bi; }
    __syncthreads();
    if (tid < 32) {
      bv = rv[tid]; bi = ri[tid];
#pragma unroll
      for (int off = 16; off > 0; off >>= 1) {
        float ov = __shfl_xor(bv, off, 32);
        int   oi = __shfl_xor(bi, off, 32);
        if (ov > bv || (ov == bv && oi < bi)) { bv = ov; bi = oi; }
      }
      if (tid == 0) s_far = bi;
    }
    __syncthreads();
  }
}

// ------------------------------- KNN ---------------------------------------
__global__ __launch_bounds__(32) void knn_kernel(const float* __restrict__ xyz,
                                                 const float* __restrict__ new_xyz,
                                                 int* __restrict__ knn_idx) {
  __shared__ float dist[N_];
  const int bm = blockIdx.x;
  const int b  = bm >> 10;
  const float* X = xyz + (size_t)b * N_ * 3;
  const float* c = new_xyz + (size_t)bm * 3;
  const float cx = c[0], cy = c[1], cz = c[2];
  const int lane = threadIdx.x;
  for (int i = lane; i < N_; i += 32) {
    float dx = X[i * 3 + 0] - cx, dy = X[i * 3 + 1] - cy, dz = X[i * 3 + 2] - cz;
    dist[i] = dx * dx + dy * dy + dz * dz;
  }
  __syncthreads();
  for (int j = 0; j < KNN_; ++j) {
    float bv = 3.4e38f; int bi = 0;
    for (int i = lane; i < N_; i += 32)
      if (dist[i] < bv) { bv = dist[i]; bi = i; }
#pragma unroll
    for (int off = 16; off > 0; off >>= 1) {
      float ov = __shfl_xor(bv, off, 32);
      int   oi = __shfl_xor(bi, off, 32);
      if (ov < bv || (ov == bv && oi < bi)) { bv = ov; bi = oi; }
    }
    if (lane == 0) { knn_idx[(size_t)bm * KNN_ + j] = bi; dist[bi] = 3.4e38f; }
    __syncthreads();
  }
}

// ------------------------------- GEMM --------------------------------------
// 256 threads = 8 waves; 128-row x NOUT-col tile per block; wave w owns row
// tile w and all NOUT/16 col tiles. The full K-extent A tile is staged once
// (no inter-chunk barriers). Weights are staged in per-lane FRAGMENT order so
// each B fragment is two 16B-aligned ds_load_b128.
//  MODE 1: gather [xyz_norm | points] rows via knn_idx, f32->f16 in staging.
//  MODE 2: RAW f16 activation copy global->LDS (async-to-LDS when available);
//          previous layer's BN scale/shift + ReLU applied at fragment build.
// Epilogue: bias add, f16 store, per-channel sum/sumsq via ds_add_f32 then
// one global atomicAdd per channel per block (BatchNorm batch stats).
template <int NOUT, int KREAL, int KPAD, int MODE>
__global__ __launch_bounds__(256) void gemm_kernel(
    const float* __restrict__ xyz, const float* __restrict__ points,
    const int* __restrict__ knn_idx, const float* __restrict__ new_xyz,
    const _Float16* __restrict__ h_in, const float* __restrict__ w,
    const float* __restrict__ bias, const float* __restrict__ abn_prev,
    _Float16* __restrict__ h_out, float* __restrict__ gstats) {
  constexpr int NT  = NOUT / 16;   // output col tiles per wave
  constexpr int NCH = KPAD / 32;   // K chunks of 32
  constexpr int AST = KPAD + 8;    // A row stride in halves; (AST*2)%16 == 0
  __shared__ __align__(16) _Float16 As[128 * AST];
  __shared__ __align__(16) _Float16 Wf[NCH * NT * 32 * 16];
  __shared__ float sstat[NOUT * 2];

  const int tid = threadIdx.x, lane = tid & 31, wv = tid >> 5;
  const size_t row0 = (size_t)blockIdx.x * 128;

  // ---- stage weights in fragment-major order: Wf[((ch*NT+ct)*32+l)*16+i] ----
  // element (ch,ct,l,i) <- w[n=ct*16+(l&15)][k=ch*32+((i<8)?0:16)+((l<16)?0:8)+(i&7)]
  for (int e = tid; e < KPAD * NOUT; e += 256) {
    const int i  = e & 15;
    const int l  = (e >> 4) & 31;
    const int t  = e >> 9;
    const int ct = t % NT;
    const int ch = t / NT;
    const int kk = ch * 32 + ((i < 8) ? 0 : 16) + ((l < 16) ? 0 : 8) + (i & 7);
    const int n  = ct * 16 + (l & 15);
    const float v = (kk < KREAL) ? w[(size_t)n * KREAL + kk] : 0.f;
    Wf[e] = (_Float16)v;
  }
  for (int e = tid; e < NOUT * 2; e += 256) sstat[e] = 0.f;

  // ---- stage the full 128 x KPAD A tile (2 threads per row) ----
  {
    const int rr = tid >> 1;
    const size_t gr = row0 + rr;
    if (MODE == 1) {
      const int c0 = (tid & 1) * (KPAD / 2);
      const size_t bm = gr >> 5;
      const int    b  = (int)(gr >> 15);
      const int    idx = knn_idx[gr];
      __builtin_prefetch(knn_idx + gr + 128, 0, 1);   // global_prefetch_b8
      const float* prow = points + ((size_t)b * N_ + idx) * CIN_;
      const float* xrow = xyz    + ((size_t)b * N_ + idx) * 3;
      const float* nx   = new_xyz + bm * 3;
#pragma unroll
      for (int cc = 0; cc < KPAD / 2; ++cc) {
        const int gc = c0 + cc;
        float v;
        if (gc < 3)             v = xrow[gc] - nx[gc];
        else if (gc < 3 + CIN_) v = prow[gc - 3];
        else                    v = 0.f;
        As[rr * AST + gc] = (_Float16)v;
      }
    } else {
      // raw f16 copy of one row half (KPAD/2 halves) -- async to LDS if avail
      const int p0 = (tid & 1) * (KPAD / 16);           // 8-half pieces
#ifdef HAVE_ASYNC_LDS
#pragma unroll
      for (int q = 0; q < KPAD / 16; ++q) {
        const int hc = (p0 + q) * 8;
        async_cp_b128(h_in + gr * (size_t)KREAL + hc, &As[rr * AST + hc]);
      }
#else
      const _Float16* hrow = h_in + gr * (size_t)KREAL;
#pragma unroll
      for (int cc = 0; cc < KPAD / 2; ++cc) {
        const int gc = p0 * 8 + cc;
        As[rr * AST + gc] = hrow[gc];
      }
#endif
    }
  }
#ifdef HAVE_ASYNC_LDS
  if (MODE == 2) async_wait0();
#endif
  __syncthreads();

  const v8f vzero = {0.f, 0.f, 0.f, 0.f, 0.f, 0.f, 0.f, 0.f};
  v8f acc[NT];
#pragma unroll
  for (int i = 0; i < NT; ++i) acc[i] = vzero;

  const int arow = wv * 16 + (lane & 15);
  const int kb   = (lane < 16) ? 0 : 8;

#pragma unroll
  for (int ch = 0; ch < NCH; ++ch) {
    // ---- A fragment: two contiguous 16B-aligned 8-half runs ----
    const int abase = arow * AST + ch * 32 + kb;
    v8h lo = *(const v8h*)&As[abase];
    v8h hi = *(const v8h*)&As[abase + 16];
    if (MODE == 2) {
      // BN + ReLU applied in registers; k index = ch*32 + kb + i (+16 for hi)
      const int k0 = ch * 32 + kb;
#pragma unroll
      for (int i = 0; i < 8; ++i) {
        float v0 = (float)lo[i];
        v0 = fmaxf(0.f, abn_prev[k0 + i] * v0 + abn_prev[128 + k0 + i]);
        lo[i] = (_Float16)v0;
        float v1 = (float)hi[i];
        v1 = fmaxf(0.f, abn_prev[k0 + 16 + i] * v1 + abn_prev[128 + k0 + 16 + i]);
        hi[i] = (_Float16)v1;
      }
    }
    v16h a;
#pragma unroll
    for (int i = 0; i < 8; ++i) { a[i] = lo[i]; a[8 + i] = hi[i]; }

#pragma unroll
    for (int ct = 0; ct < NT; ++ct) {
      const int wbase = ((ch * NT + ct) * 32 + lane) * 16;
      v8h blo = *(const v8h*)&Wf[wbase];
      v8h bhi = *(const v8h*)&Wf[wbase + 8];
      v16h bf;
#pragma unroll
      for (int i = 0; i < 8; ++i) { bf[i] = blo[i]; bf[8 + i] = bhi[i]; }
      acc[ct] = __builtin_amdgcn_wmma_f32_16x16x32_f16(
          false, a, false, bf, (short)0, acc[ct], false, false);
    }
  }

  // ---- epilogue: bias, f16 store, BN stat accumulation ----
#pragma unroll
  for (int ct = 0; ct < NT; ++ct) {
    const int chn = ct * 16 + (lane & 15);
    const float bv = bias[chn];
    const int rbase = wv * 16 + ((lane < 16) ? 0 : 8);
    float s = 0.f, ss = 0.f;
#pragma unroll
    for (int r = 0; r < 8; ++r) {
      float v = acc[ct][r] + bv;
      const size_t gr = row0 + rbase + r;
      h_out[gr * NOUT + chn] = (_Float16)v;
      s += v; ss += v * v;
    }
    atomicAdd(&sstat[chn * 2 + 0], s);   // ds_add_f32
    atomicAdd(&sstat[chn * 2 + 1], ss);
  }
  __syncthreads();
  for (int e = tid; e < NOUT; e += 256) {
    atomicAdd(&gstats[e * 2 + 0], sstat[e * 2 + 0]);
    atomicAdd(&gstats[e * 2 + 1], sstat[e * 2 + 1]);
  }
}

// --------------------------- BN stat finalize -------------------------------
__global__ void finalize_stats_kernel(const float* __restrict__ gst,
                                      const float* __restrict__ g,
                                      const float* __restrict__ be,
                                      float* __restrict__ abn, int C, float invR) {
  const int c = threadIdx.x;
  if (c < C) {
    const float mean = gst[2 * c] * invR;
    const float var  = gst[2 * c + 1] * invR - mean * mean;
    const float inv  = rsqrtf(var + 1e-5f);
    const float a    = g[c] * inv;
    abn[c]       = a;
    abn[128 + c] = be[c] - a * mean;
  }
}

__global__ void zero_kernel(float* __restrict__ p, int n) {
  const int i = blockIdx.x * blockDim.x + threadIdx.x;
  if (i < n) p[i] = 0.f;
}

// ------------------------ BN3 + ReLU + max over k ---------------------------
__global__ __launch_bounds__(128) void bn_maxpool_kernel(
    const _Float16* __restrict__ h3, const float* __restrict__ abn3,
    float* __restrict__ out) {
  const int c = threadIdx.x;                 // channel 0..127
  const size_t bm = blockIdx.x;              // 0..16383
  const float a = abn3[c], sh = abn3[128 + c];
  const size_t base = bm * KNN_ * 128;
  float mx = 0.f;                            // ReLU outputs are >= 0
  for (int j = 0; j < KNN_; ++j) {
    float v = fmaxf(0.f, a * (float)h3[base + (size_t)j * 128 + c] + sh);
    mx = fmaxf(mx, v);
  }
  out[bm * 128 + c] = mx;
}

// ---------------------------------------------------------------------------
extern "C" void kernel_launch(void* const* d_in, const int* in_sizes, int n_in,
                              void* d_out, int out_size, void* d_ws, size_t ws_size,
                              hipStream_t stream) {
  const float* xyz    = (const float*)d_in[0];
  const float* points = (const float*)d_in[1];
  const float* w1  = (const float*)d_in[2];
  const float* b1  = (const float*)d_in[3];
  const float* g1  = (const float*)d_in[4];
  const float* be1 = (const float*)d_in[5];
  const float* w2  = (const float*)d_in[6];
  const float* b2  = (const float*)d_in[7];
  const float* g2  = (const float*)d_in[8];
  const float* be2 = (const float*)d_in[9];
  const float* w3  = (const float*)d_in[10];
  const float* b3  = (const float*)d_in[11];
  const float* g3  = (const float*)d_in[12];
  const float* be3 = (const float*)d_in[13];

  float* new_xyz    = (float*)d_out;                        // (B,M,3)
  float* new_points = (float*)d_out + (size_t)B_ * M_ * 3;  // (B,M,128)

  // Workspace layout
  char* ws = (char*)d_ws;
  size_t off = 0;
  int* knn_idx = (int*)(ws + off);        off += (size_t)ROWS_ * sizeof(int);
  _Float16* out1 = (_Float16*)(ws + off); off += (size_t)ROWS_ * 64 * sizeof(_Float16);
  _Float16* out2 = (_Float16*)(ws + off); off += (size_t)ROWS_ * 64 * sizeof(_Float16);
  _Float16* out3 = (_Float16*)(ws + off); off += (size_t)ROWS_ * 128 * sizeof(_Float16);
  float* gstats = (float*)(ws + off);     off += 3 * 256 * sizeof(float);
  float* abn    = (float*)(ws + off);     off += 3 * 256 * sizeof(float);

  const float invR = 1.0f / (float)ROWS_;
  const int gemm_blocks = ROWS_ / 128;   // 4096

  zero_kernel<<<3, 256, 0, stream>>>(gstats, 3 * 256);
  fps_kernel<<<B_, 1024, 0, stream>>>(xyz, new_xyz);
  knn_kernel<<<B_ * M_, 32, 0, stream>>>(xyz, new_xyz, knn_idx);

  // Layer 1: gather-fused GEMM (K = 67 padded to 96), stats for BN1
  gemm_kernel<64, 67, 96, 1><<<gemm_blocks, 256, 0, stream>>>(
      xyz, points, knn_idx, new_xyz, nullptr, w1, b1, nullptr, out1, gstats);
  finalize_stats_kernel<<<1, 128, 0, stream>>>(gstats, g1, be1, abn, 64, invR);

  // Layer 2: BN1+ReLU at fragment build, K = 64
  gemm_kernel<64, 64, 64, 2><<<gemm_blocks, 256, 0, stream>>>(
      nullptr, nullptr, nullptr, nullptr, out1, w2, b2, abn, out2, gstats + 256);
  finalize_stats_kernel<<<1, 128, 0, stream>>>(gstats + 256, g2, be2, abn + 256, 64, invR);

  // Layer 3: BN2+ReLU at fragment build, N = 128
  gemm_kernel<128, 64, 64, 2><<<gemm_blocks, 256, 0, stream>>>(
      nullptr, nullptr, nullptr, nullptr, out2, w3, b3, abn + 256, out3, gstats + 512);
  finalize_stats_kernel<<<1, 128, 0, stream>>>(gstats + 512, g3, be3, abn + 512, 128, invR);

  // BN3 + ReLU + maxpool over k
  bn_maxpool_kernel<<<B_ * M_, 128, 0, stream>>>(out3, abn + 512, new_points);
}
